// Model_2456721293382
// MI455X (gfx1250) — matrix operand order repaired and checked by reading, not compiled, not run on hardware
//
#include <hip/hip_runtime.h>
#include <cstdint>
#include <cstddef>

// ---------------- CDNA5 WMMA types ----------------
typedef __attribute__((ext_vector_type(16))) __bf16       v16bf;
typedef __attribute__((ext_vector_type(8)))  float        v8f;
typedef __attribute__((ext_vector_type(4)))  unsigned int u32x4;

union Frag16 { v16bf bf; u32x4 q[2]; };

__device__ __forceinline__ unsigned short f2bf(float f) {
  unsigned int u = __float_as_uint(f);
  return (unsigned short)((u + 0x7fffu + ((u >> 16) & 1u)) >> 16); // RNE
}

// ---------------- elementwise / utility kernels ----------------
__global__ void k_zero(float* __restrict__ p, size_t n) {
  for (size_t i = (size_t)blockIdx.x * blockDim.x + threadIdx.x; i < n;
       i += (size_t)gridDim.x * blockDim.x)
    p[i] = 0.0f;
}

__global__ void k_scatter_avg(const float* __restrict__ feats, const int* __restrict__ p2v,
                              float* __restrict__ s, float* __restrict__ cnt, int N) {
  int i = blockIdx.x * blockDim.x + threadIdx.x;
  if (i >= N) return;
  int v = p2v[i];
  atomicAdd(&s[(size_t)v * 3 + 0], feats[(size_t)i * 3 + 0]);
  atomicAdd(&s[(size_t)v * 3 + 1], feats[(size_t)i * 3 + 1]);
  atomicAdd(&s[(size_t)v * 3 + 2], feats[(size_t)i * 3 + 2]);
  atomicAdd(&cnt[v], 1.0f);
}

__global__ void k_divide_avg(float* __restrict__ s, const float* __restrict__ cnt, int V) {
  int i = blockIdx.x * blockDim.x + threadIdx.x;
  if (i >= V) return;
  float c = fmaxf(cnt[i], 1.0f);
  s[(size_t)i * 3 + 0] /= c;
  s[(size_t)i * 3 + 1] /= c;
  s[(size_t)i * 3 + 2] /= c;
}

// stem: 27-tap subconv, Cin=3 -> Cout=32, fp32 VALU (K too small for WMMA)
__global__ void k_stem(const float* __restrict__ vx, const int* __restrict__ nbr,
                       const float* __restrict__ w, float* __restrict__ out, int V) {
  int i = blockIdx.x * blockDim.x + threadIdx.x;
  if (i >= V * 32) return;
  int v = i >> 5, co = i & 31;
  float a = 0.0f;
  for (int k = 0; k < 27; ++k) {
    int r = nbr[(size_t)v * 27 + k];
    if (r < V) {
      const float* xr = vx + (size_t)r * 3;
      const float* wr = w + (size_t)k * 96 + co;
      a += xr[0] * wr[0] + xr[1] * wr[32] + xr[2] * wr[64];
    }
  }
  out[i] = a;
}

// pack f32 weights [27][Cin][Cout] into per-lane bf16 B fragments:
// wp[(((k*Cin/32+cic)*(Cout/16)+coc)*32 + lane)*16 + j] = w[k][cic*32+16*(lane/16)+j][coc*16+lane%16]
__global__ void k_pack_w(const float* __restrict__ w, unsigned short* __restrict__ wp,
                         int Cin, int Cout) {
  int n = 27 * Cin * Cout;
  int i = blockIdx.x * blockDim.x + threadIdx.x;
  if (i >= n) return;
  int k  = i / (Cin * Cout);
  int r  = i - k * Cin * Cout;
  int ci = r / Cout;
  int co = r - ci * Cout;
  int cic = ci >> 5, cir = ci & 31, half = cir >> 4, j = cir & 15;
  int coc = co >> 4, nn = co & 15, lane = (half << 4) | nn;
  size_t dst = ((((size_t)k * (Cin >> 5) + cic) * (Cout >> 4) + coc) * 32 + lane) * 16 + j;
  wp[dst] = f2bf(w[i]);
}

// ---------------- batchnorm ----------------
__global__ __launch_bounds__(256) void k_bn_stats(const float* __restrict__ x, int V, int C,
                                                  int ld, float* __restrict__ stats) {
  int c = blockIdx.x; // one block per channel
  float s = 0.0f, ss = 0.0f;
  for (int v = threadIdx.x; v < V; v += 256) {
    float t = x[(size_t)v * ld + c];
    s += t; ss += t * t;
  }
  __shared__ float sh[512];
  sh[threadIdx.x] = s; sh[256 + threadIdx.x] = ss;
  __syncthreads();
  for (int o = 128; o > 0; o >>= 1) {
    if (threadIdx.x < (unsigned)o) {
      sh[threadIdx.x] += sh[threadIdx.x + o];
      sh[256 + threadIdx.x] += sh[256 + threadIdx.x + o];
    }
    __syncthreads();
  }
  if (threadIdx.x == 0) { stats[c] = sh[0]; stats[C + c] = sh[256]; }
}

__global__ void k_bn_coef(const float* __restrict__ stats, const float* __restrict__ g,
                          const float* __restrict__ b, int V, int C, float* __restrict__ coef) {
  int c = blockIdx.x * blockDim.x + threadIdx.x;
  if (c >= C) return;
  float inv = 1.0f / (float)V;
  float mu = stats[c] * inv;
  float var = stats[C + c] * inv - mu * mu;
  float sc = g[c] * rsqrtf(var + 1e-4f);
  coef[c] = sc;
  coef[C + c] = b[c] - mu * sc;
}

__global__ void k_bnrelu_f32(const float* __restrict__ x, const float* __restrict__ coef,
                             float* __restrict__ y, int V, int C, int ldi, int ldo) {
  size_t n = (size_t)V * C;
  size_t i = (size_t)blockIdx.x * blockDim.x + threadIdx.x;
  if (i >= n) return;
  int v = (int)(i / C), c = (int)(i - (size_t)v * C);
  float t = x[(size_t)v * ldi + c] * coef[c] + coef[C + c];
  y[(size_t)v * ldo + c] = fmaxf(t, 0.0f);
}

// writes (V+1) dense rows of bf16; row V is the zero "missing neighbor" pad row
__global__ void k_bnrelu_bf16(const float* __restrict__ x, const float* __restrict__ coef,
                              unsigned short* __restrict__ y, int V, int C, int ldi) {
  size_t n = (size_t)(V + 1) * C;
  size_t i = (size_t)blockIdx.x * blockDim.x + threadIdx.x;
  if (i >= n) return;
  int v = (int)(i / C), c = (int)(i - (size_t)v * C);
  unsigned short o = 0;
  if (v < V) {
    float t = x[(size_t)v * ldi + c] * coef[c] + coef[C + c];
    o = f2bf(fmaxf(t, 0.0f));
  }
  y[i] = o;
}

// ---------------- 27-offset submanifold conv on WMMA (bf16 -> f32) ----------------
// one wave32 computes a 16-voxel x 32-cout tile; A reused across two B tiles.
// grid = (voxel-tile groups, cout-pair tiles); templated on Cin/32 for full unroll.
template <int CCH>
__global__ __launch_bounds__(256) void k_subconv_wmma(
    const unsigned short* __restrict__ xbf,  // (V+1) x Cin bf16, row V = zeros
    const int* __restrict__ nbr,             // V x 27
    const unsigned short* __restrict__ wp,   // packed B fragments
    float* __restrict__ out, int V, int Cout, int ldo, int coff) {
  const int nVT = (V + 15) >> 4;
  const int vt = blockIdx.x * (blockDim.x >> 5) + (threadIdx.x >> 5);
  if (vt >= nVT) return;                     // wave-uniform exit, EXEC stays full
  const int co = blockIdx.y << 5;
  const int lane = threadIdx.x & 31;
  const int m = lane & 15;
  const int half = lane >> 4;
  const int vrow = vt * 16 + m;
  const int Cin = CCH * 32;
  const int cochunks = Cout >> 4;
  const size_t cstride = (size_t)cochunks * 512;  // ushorts per cic step
  const size_t kstride = (size_t)CCH * cstride;   // ushorts per k step
  const unsigned short* wbase = wp + (size_t)(co >> 4) * 512 + (size_t)lane * 16;
  v8f acc0 = {0.f, 0.f, 0.f, 0.f, 0.f, 0.f, 0.f, 0.f};
  v8f acc1 = acc0;
  int r = (vrow < V) ? nbr[(size_t)vrow * 27] : V;  // V => zero pad row
  for (int k = 0; k < 27; ++k) {
    const unsigned short* xrow = xbf + (size_t)r * Cin;
    int rn = V;
    if (k < 26) rn = (vrow < V) ? nbr[(size_t)vrow * 27 + k + 1] : V;
    __builtin_prefetch(xbf + (size_t)rn * Cin, 0, 0);  // global_prefetch of next gather row
    const unsigned short* wk = wbase + (size_t)k * kstride;
#pragma unroll
    for (int cic = 0; cic < CCH; ++cic) {
      Frag16 a, b0, b1;
      const unsigned short* ap = xrow + (cic << 5) + (half << 4);
      a.q[0] = *(const u32x4*)(ap);         // K = 16*half .. +7
      a.q[1] = *(const u32x4*)(ap + 8);     // K = 16*half+8 .. +7
      const unsigned short* bp = wk + (size_t)cic * cstride;
      b0.q[0] = *(const u32x4*)(bp);
      b0.q[1] = *(const u32x4*)(bp + 8);
      b1.q[0] = *(const u32x4*)(bp + 512);  // next 16-wide cout tile
      b1.q[1] = *(const u32x4*)(bp + 520);
      acc0 = __builtin_amdgcn_wmma_f32_16x16x32_bf16(false, a.bf, false, b0.bf,
                                                     (short)0, acc0, false, false);
      acc1 = __builtin_amdgcn_wmma_f32_16x16x32_bf16(false, a.bf, false, b1.bf,
                                                     (short)0, acc1, false, false);
    }
    r = rn;
  }
  // D layout: VGPR i -> M = 8*half + i, N = lane%16
  const int nn = lane & 15;
#pragma unroll
  for (int i = 0; i < 8; ++i) {
    int row = vt * 16 + (half << 3) + i;
    if (row < V) {
      float* orow = out + (size_t)row * ldo + coff + co;
      orow[nn]      = acc0[i];
      orow[16 + nn] = acc1[i];
    }
  }
}

// ---------------- strided 2x2x2 down / up convolutions (fp32 VALU) ----------------
__global__ void k_down(const float* __restrict__ h, const int* __restrict__ parent,
                       const int* __restrict__ offv, const float* __restrict__ w,
                       float* __restrict__ y, int V, int nP, int nQ) {
  size_t n = (size_t)V * nQ;
  size_t i = (size_t)blockIdx.x * blockDim.x + threadIdx.x;
  if (i >= n) return;
  int v = (int)(i / nQ), co = (int)(i - (size_t)v * nQ);
  const float* wr = w + ((size_t)offv[v] * nP) * nQ + co;
  const float* hr = h + (size_t)v * nP;
  float a = 0.0f;
  for (int c = 0; c < nP; ++c) a += hr[c] * wr[(size_t)c * nQ];
  atomicAdd(&y[(size_t)parent[v] * nQ + co], a);
}

__global__ void k_up(const float* __restrict__ hc, const int* __restrict__ parent,
                     const int* __restrict__ offv, const float* __restrict__ w,
                     float* __restrict__ out, int ldo, int coff, int V, int nQ, int nP) {
  size_t n = (size_t)V * nP;
  size_t i = (size_t)blockIdx.x * blockDim.x + threadIdx.x;
  if (i >= n) return;
  int v = (int)(i / nP), co = (int)(i - (size_t)v * nP);
  const float* wr = w + ((size_t)offv[v] * nQ) * nP + co;
  const float* hr = hc + (size_t)parent[v] * nQ;
  float a = 0.0f;
  for (int c = 0; c < nQ; ++c) a += hr[c] * wr[(size_t)c * nP];
  out[(size_t)v * ldo + coff + co] = a;
}

// ---------------- output head: gather voxel->point, 32->20 linear ----------------
__global__ void k_head(const float* __restrict__ h, const int* __restrict__ p2v,
                       const float* __restrict__ wl, const float* __restrict__ bl,
                       float* __restrict__ out, int N) {
  size_t n = (size_t)N * 20;
  size_t i = (size_t)blockIdx.x * blockDim.x + threadIdx.x;
  if (i >= n) return;
  int p = (int)(i / 20), j = (int)(i - (size_t)p * 20);
  const float* hr = h + (size_t)p2v[p] * 32;
  float a = bl[j];
  for (int c = 0; c < 32; ++c) a += hr[c] * wl[c * 20 + j];
  out[i] = a;
}

// ---------------- host orchestration ----------------
static const int PL[7] = {32, 64, 96, 128, 160, 192, 224};

struct Ctx {
  hipStream_t st;
  int Vl[7];
  const int *nbr[7], *parent[6], *offv[6];
  const float *g_pre[7], *b_pre[7];
  const float *g_down[6], *b_down[6], *w_down[6];
  const float *g_up[6], *b_up[6], *w_up[6];
  const float *g_post[6], *b_post[6];
  unsigned short *wp_pre[7], *wp_post[6];
  float *concat[6], *ybuf[6], *xout[7];
  void *scratch[7];
  float *stats, *coef;
};

static inline unsigned cdiv(size_t a, unsigned b) { return (unsigned)((a + b - 1) / b); }

static void bn(Ctx& C, const float* x, int V, int Cc, int ld, const float* g, const float* b) {
  k_bn_stats<<<Cc, 256, 0, C.st>>>(x, V, Cc, ld, C.stats);
  k_bn_coef<<<cdiv(Cc, 64), 64, 0, C.st>>>(C.stats, g, b, V, Cc, C.coef);
}

static void subconv(Ctx& C, const unsigned short* xbf, const int* nbr,
                    const unsigned short* wp, float* out, int V, int Cin, int Cout,
                    int ldo, int coff) {
  dim3 grid(cdiv((size_t)(V + 15) / 16, 8), (unsigned)(Cout / 32), 1);
  switch (Cin / 32) {
#define SC_CASE(N)                                                                     \
  case N:                                                                              \
    k_subconv_wmma<N><<<grid, 256, 0, C.st>>>(xbf, nbr, wp, out, V, Cout, ldo, coff);  \
    break;
    SC_CASE(1) SC_CASE(2) SC_CASE(3) SC_CASE(4) SC_CASE(5) SC_CASE(6) SC_CASE(7)
    SC_CASE(8) SC_CASE(10) SC_CASE(12) SC_CASE(14)
#undef SC_CASE
    default: break;  // channel counts in this net are always one of the above
  }
}

static float* run_level(Ctx& C, int l, const float* xin) {
  const int V = C.Vl[l], nP = PL[l];
  // pre: BNReLU -> subconv (bf16 WMMA)
  bn(C, xin, V, nP, nP, C.g_pre[l], C.b_pre[l]);
  unsigned short* hbf = (unsigned short*)C.scratch[l];
  k_bnrelu_bf16<<<cdiv((size_t)(V + 1) * nP, 256), 256, 0, C.st>>>(xin, C.coef, hbf, V, nP, nP);
  if (l == 6) {
    subconv(C, hbf, C.nbr[6], C.wp_pre[6], C.xout[6], V, nP, nP, nP, 0);
    return C.xout[6];
  }
  const int nQ = PL[l + 1], Vc = C.Vl[l + 1];
  subconv(C, hbf, C.nbr[l], C.wp_pre[l], C.concat[l], V, nP, nP, 2 * nP, 0);
  // down path
  bn(C, C.concat[l], V, nP, 2 * nP, C.g_down[l], C.b_down[l]);
  float* hdn = (float*)C.scratch[l];
  k_bnrelu_f32<<<cdiv((size_t)V * nP, 256), 256, 0, C.st>>>(C.concat[l], C.coef, hdn, V, nP, 2 * nP, nP);
  k_zero<<<cdiv((size_t)Vc * nQ, 256), 256, 0, C.st>>>(C.ybuf[l], (size_t)Vc * nQ);
  k_down<<<cdiv((size_t)V * nQ, 256), 256, 0, C.st>>>(hdn, C.parent[l], C.offv[l], C.w_down[l],
                                                     C.ybuf[l], V, nP, nQ);
  float* xc = run_level(C, l + 1, C.ybuf[l]);
  // up path
  bn(C, xc, Vc, nQ, nQ, C.g_up[l], C.b_up[l]);
  float* hup = (float*)C.scratch[l];
  k_bnrelu_f32<<<cdiv((size_t)Vc * nQ, 256), 256, 0, C.st>>>(xc, C.coef, hup, Vc, nQ, nQ, nQ);
  k_up<<<cdiv((size_t)V * nP, 256), 256, 0, C.st>>>(hup, C.parent[l], C.offv[l], C.w_up[l],
                                                   C.concat[l], 2 * nP, nP, V, nQ, nP);
  // post: BNReLU(2nP) -> subconv(2nP -> nP)
  bn(C, C.concat[l], V, 2 * nP, 2 * nP, C.g_post[l], C.b_post[l]);
  unsigned short* hbf2 = (unsigned short*)C.scratch[l];
  k_bnrelu_bf16<<<cdiv((size_t)(V + 1) * 2 * nP, 256), 256, 0, C.st>>>(C.concat[l], C.coef, hbf2,
                                                                      V, 2 * nP, 2 * nP);
  subconv(C, hbf2, C.nbr[l], C.wp_post[l], C.xout[l], V, 2 * nP, nP, nP, 0);
  return C.xout[l];
}

extern "C" void kernel_launch(void* const* d_in, const int* in_sizes, int n_in,
                              void* d_out, int out_size, void* d_ws, size_t ws_size,
                              hipStream_t stream) {
  Ctx C;
  C.st = stream;
  // ---- input map (flattened setup_inputs dict order) ----
  const float* feats = (const float*)d_in[0];
  const float* w_in  = (const float*)d_in[1];
  const float *w_pre[7], *w_post[6];
  for (int l = 0; l < 6; ++l) {
    int base = 2 + 12 * l;
    C.g_pre[l]  = (const float*)d_in[base + 0];
    C.b_pre[l]  = (const float*)d_in[base + 1];
    w_pre[l]    = (const float*)d_in[base + 2];
    C.g_down[l] = (const float*)d_in[base + 3];
    C.b_down[l] = (const float*)d_in[base + 4];
    C.w_down[l] = (const float*)d_in[base + 5];
    C.g_up[l]   = (const float*)d_in[base + 6];
    C.b_up[l]   = (const float*)d_in[base + 7];
    C.w_up[l]   = (const float*)d_in[base + 8];
    C.g_post[l] = (const float*)d_in[base + 9];
    C.b_post[l] = (const float*)d_in[base + 10];
    w_post[l]   = (const float*)d_in[base + 11];
  }
  C.g_pre[6] = (const float*)d_in[74];
  C.b_pre[6] = (const float*)d_in[75];
  w_pre[6]   = (const float*)d_in[76];
  const float* g_out = (const float*)d_in[77];
  const float* b_out = (const float*)d_in[78];
  const float* w_lin = (const float*)d_in[79];
  const float* b_lin = (const float*)d_in[80];
  for (int l = 0; l < 6; ++l) {
    C.nbr[l]    = (const int*)d_in[81 + 3 * l];
    C.parent[l] = (const int*)d_in[82 + 3 * l];
    C.offv[l]   = (const int*)d_in[83 + 3 * l];
    C.Vl[l]     = in_sizes[81 + 3 * l] / 27;
  }
  C.nbr[6] = (const int*)d_in[99];
  C.Vl[6]  = in_sizes[99] / 27;
  const int* p2v = (const int*)d_in[100];
  const int Np = in_sizes[0] / 3;
  (void)n_in; (void)out_size; (void)ws_size;

  // ---- workspace bump allocator ----
  char* base = (char*)d_ws;
  size_t off = 0;
  auto alloc = [&](size_t bytes) -> void* {
    off = (off + 255) & ~(size_t)255;
    void* p = base + off;
    off += bytes;
    return p;
  };
  const int V0 = C.Vl[0];
  float* vsum = (float*)alloc((size_t)V0 * 3 * 4);
  float* vcnt = (float*)alloc((size_t)V0 * 4);
  float* x0   = (float*)alloc((size_t)V0 * 32 * 4);
  C.stats = (float*)alloc(1024 * 4);
  C.coef  = (float*)alloc(1024 * 4);
  for (int l = 0; l < 7; ++l) {
    int nP = PL[l];
    C.wp_pre[l] = (unsigned short*)alloc((size_t)27 * nP * nP * 2);
    if (l < 6) C.wp_post[l] = (unsigned short*)alloc((size_t)27 * 2 * nP * nP * 2);
  }
  for (int l = 0; l < 7; ++l) {
    size_t V = (size_t)C.Vl[l];
    int nP = PL[l];
    if (l < 6) {
      size_t Vc = (size_t)C.Vl[l + 1];
      int nQ = PL[l + 1];
      C.concat[l] = (float*)alloc(V * 2 * nP * 4);
      size_t sc = V * nP * 4;                               // hdn
      size_t t1 = (V + 1) * 2 * (size_t)nP * 2;             // hbf_post
      size_t t2 = Vc * (size_t)nQ * 4;                      // hup
      size_t t3 = (V + 1) * (size_t)nP * 2;                 // hbf_pre
      if (t1 > sc) sc = t1;
      if (t2 > sc) sc = t2;
      if (t3 > sc) sc = t3;
      C.scratch[l] = alloc(sc);
      C.ybuf[l] = (float*)alloc(Vc * (size_t)nQ * 4);
      C.xout[l] = (float*)alloc(V * (size_t)nP * 4);
    } else {
      C.scratch[6] = alloc((V + 1) * (size_t)nP * 2);
      C.xout[6]    = (float*)alloc(V * (size_t)nP * 4);
    }
  }

  // ---- weight pre-pack to bf16 B-fragment layout (idempotent per call) ----
  for (int l = 0; l < 7; ++l) {
    int nP = PL[l];
    k_pack_w<<<cdiv((size_t)27 * nP * nP, 256), 256, 0, stream>>>(w_pre[l], C.wp_pre[l], nP, nP);
    if (l < 6)
      k_pack_w<<<cdiv((size_t)27 * 2 * nP * nP, 256), 256, 0, stream>>>(w_post[l], C.wp_post[l],
                                                                       2 * nP, nP);
  }

  // ---- input layer: average point features per voxel ----
  k_zero<<<cdiv((size_t)V0 * 3, 256), 256, 0, stream>>>(vsum, (size_t)V0 * 3);
  k_zero<<<cdiv((size_t)V0, 256), 256, 0, stream>>>(vcnt, (size_t)V0);
  k_scatter_avg<<<cdiv((size_t)Np, 256), 256, 0, stream>>>(feats, p2v, vsum, vcnt, Np);
  k_divide_avg<<<cdiv((size_t)V0, 256), 256, 0, stream>>>(vsum, vcnt, V0);
  // ---- stem conv 3 -> 32 ----
  k_stem<<<cdiv((size_t)V0 * 32, 256), 256, 0, stream>>>(vsum, C.nbr[0], w_in, x0, V0);

  // ---- U-Net ----
  float* xf = run_level(C, 0, x0);

  // ---- output head ----
  bn(C, xf, V0, 32, 32, g_out, b_out);
  float* hf = (float*)C.scratch[0];
  k_bnrelu_f32<<<cdiv((size_t)V0 * 32, 256), 256, 0, stream>>>(xf, C.coef, hf, V0, 32, 32, 32);
  k_head<<<cdiv((size_t)Np * 20, 256), 256, 0, stream>>>(hf, p2v, w_lin, b_lin, (float*)d_out, Np);
}
